// KMeans_27487790695165
// MI455X (gfx1250) — matrix operand order repaired and checked by reading, not compiled, not run on hardware
//
#include <hip/hip_runtime.h>

// ---- CDNA5 (gfx1250) types ----
typedef __bf16 bf16_t;
typedef __attribute__((ext_vector_type(16))) __bf16 v16bf;
typedef __attribute__((ext_vector_type(8)))  float  v8f;

#define N_POINTS   65536
#define N_CLUSTERS 4096
#define N_FEAT     512
#define M_TILE     128   // rows of x per block
#define KC_TILE    128   // clusters per LDS chunk
#define PITCH      520   // bf16 elems per LDS row (512 + 8 pad -> conflict-free b128)

// LDS layout (bytes)
#define XS_OFF     0                      // 128*520*2 = 133120
#define CS_OFF     133120                 // 128*520*2 = 133120
#define CN_OFF     266240                 // 128*4 = 512
#define PT_OFF     266752                 // 256*4 = 1024
#define LDS_BYTES  267776                 // ~261.5 KB < 320 KB/WGP

union FragU { uint4 q[2]; v16bf v; };

__global__ __launch_bounds__(256)
void kmeans_assign_wmma(const float* __restrict__ x,
                        const float* __restrict__ centers,
                        int* __restrict__ out) {
  extern __shared__ char smem[];
  bf16_t* Xs     = (bf16_t*)(smem + XS_OFF);
  bf16_t* Cs     = (bf16_t*)(smem + CS_OFF);
  float*  cnormS = (float*)(smem + CN_OFF);
  float*  partS  = (float*)(smem + PT_OFF);

  const int t    = threadIdx.x;
  const int blk  = blockIdx.x;
  const int r0   = blk * M_TILE;

  const int lane = t & 31;        // wave32
  const int wv   = t >> 5;        // 0..7
  const int wr   = wv >> 1;       // 0..3 : 32-row group (2 stripes of 16)
  const int wc   = wv & 1;        // 0..1 : 64-col group (4 tiles of 16)
  const int h    = lane >> 4;     // half-wave (K-split for A/B, M-split for D)
  const int n    = lane & 15;     // A: row-in-stripe; B/D: column-in-tile

  // ---- load & convert the 128x512 x-block to bf16 in LDS ----
  {
    const float4* xg = (const float4*)(x + (size_t)r0 * N_FEAT);
    #pragma unroll 4
    for (int i = 0; i < 64; ++i) {
      int flat4 = i * 256 + t;            // 0..16383 float4s of the block
      int row   = flat4 >> 7;             // /128 float4 per row
      int col4  = flat4 & 127;
      float4 v  = xg[flat4];
      union { bf16_t b[4]; uint2 u; } pk;
      pk.b[0] = (bf16_t)v.x; pk.b[1] = (bf16_t)v.y;
      pk.b[2] = (bf16_t)v.z; pk.b[3] = (bf16_t)v.w;
      *(uint2*)(Xs + row * PITCH + col4 * 4) = pk.u;
    }
  }

  float best[16];
  int   bidx[16];
  #pragma unroll
  for (int r = 0; r < 16; ++r) { best[r] = 3.4e38f; bidx[r] = 0; }

  // A-fragment bases: stripe rows 32*wr + 16*s + n, half-wave K offset 8*h
  const bf16_t* ap0 = Xs + (32 * wr + n) * PITCH + 8 * h;
  const bf16_t* ap1 = ap0 + 16 * PITCH;
  // B-fragment base: row = 64*wc + 16*c + n, half-wave K offset 8*h
  const bf16_t* bp  = Cs + (64 * wc + n) * PITCH + 8 * h;

  for (int kc = 0; kc < N_CLUSTERS / KC_TILE; ++kc) {
    __syncthreads();   // protect Cs/cnormS from previous chunk's readers

    // ---- convert 128x512 center chunk to bf16 + fp32 row norms ----
    {
      const int row = t >> 1;       // 0..127 cluster row in chunk
      const int seg = t & 1;        // 256-float segment
      const float* cg = centers + (size_t)(kc * KC_TILE + row) * N_FEAT + seg * 256;
      bf16_t* cl = Cs + row * PITCH + seg * 256;
      float s = 0.f;
      #pragma unroll 4
      for (int i = 0; i < 64; ++i) {
        float4 v = ((const float4*)cg)[i];
        s = fmaf(v.x, v.x, fmaf(v.y, v.y, fmaf(v.z, v.z, fmaf(v.w, v.w, s))));
        union { bf16_t b[4]; uint2 u; } pk;
        pk.b[0] = (bf16_t)v.x; pk.b[1] = (bf16_t)v.y;
        pk.b[2] = (bf16_t)v.z; pk.b[3] = (bf16_t)v.w;
        *(uint2*)(cl + i * 4) = pk.u;
      }
      partS[t] = s;
      if (kc + 1 < N_CLUSTERS / KC_TILE)   // warm L2 for next chunk (global_prefetch_b8)
        __builtin_prefetch(centers + (size_t)((kc + 1) * KC_TILE + row) * N_FEAT + seg * 256, 0, 1);
    }
    __syncthreads();
    if (t < KC_TILE)
      cnormS[t] = partS[2 * t] + partS[2 * t + 1];
    __syncthreads();

    // ---- 128x128 += Xblk(128x512) * Cchunk^T(512x128), bf16 WMMA, f32 acc ----
    // per wave: 2 row-stripes x 4 col-tiles -> B reused across 2 WMMAs
    v8f acc[2][4];
    #pragma unroll
    for (int c = 0; c < 4; ++c) { acc[0][c] = (v8f){}; acc[1][c] = (v8f){}; }

    #pragma unroll 4
    for (int f = 0; f < N_FEAT; f += 32) {
      FragU a0, a1;
      a0.q[0] = *(const uint4*)(ap0 + f);
      a0.q[1] = *(const uint4*)(ap0 + f + 16);
      a1.q[0] = *(const uint4*)(ap1 + f);
      a1.q[1] = *(const uint4*)(ap1 + f + 16);
      #pragma unroll
      for (int c = 0; c < 4; ++c) {
        FragU b;
        b.q[0] = *(const uint4*)(bp + c * 16 * PITCH + f);
        b.q[1] = *(const uint4*)(bp + c * 16 * PITCH + f + 16);
        acc[0][c] = __builtin_amdgcn_wmma_f32_16x16x32_bf16(false, a0.v, false, b.v, (short)0, acc[0][c], false, false);
        acc[1][c] = __builtin_amdgcn_wmma_f32_16x16x32_bf16(false, a1.v, false, b.v, (short)0, acc[1][c], false, false);
      }
    }

    // ---- fused argmin epilogue: dist_k = c_norm[k] - 2*s (||x||^2 drops out) ----
    const int kbase = kc * KC_TILE + 64 * wc + n;
    #pragma unroll
    for (int c = 0; c < 4; ++c) {
      const float cn = cnormS[64 * wc + 16 * c + n];
      const int   id = kbase + 16 * c;
      #pragma unroll
      for (int s = 0; s < 2; ++s) {
        #pragma unroll
        for (int j = 0; j < 8; ++j) {   // D layout: VGPR j, lane 16h+n -> row 8h+j, col n
          float d = fmaf(-2.f, acc[s][c][j], cn);
          int   r = s * 8 + j;
          if (d < best[r]) { best[r] = d; bidx[r] = id; }
        }
      }
    }
  }

  // ---- cross-lane min over the 16 column-lanes of each half-wave ----
  #pragma unroll
  for (int r = 0; r < 16; ++r) {
    float b = best[r];
    int   id = bidx[r];
    #pragma unroll
    for (int off = 8; off >= 1; off >>= 1) {
      float ob = __shfl_xor(b, off, 32);
      int   oi = __shfl_xor(id, off, 32);
      if (ob < b || (ob == b && oi < id)) { b = ob; id = oi; }
    }
    if (n == 0)   // row = 32*wr + 16*s + 8h + j,  r = s*8 + j
      out[r0 + 32 * wr + 16 * (r >> 3) + 8 * h + (r & 7)] = id;
  }
}

extern "C" void kernel_launch(void* const* d_in, const int* in_sizes, int n_in,
                              void* d_out, int out_size, void* d_ws, size_t ws_size,
                              hipStream_t stream) {
  (void)in_sizes; (void)n_in; (void)out_size; (void)d_ws; (void)ws_size;
  const float* x       = (const float*)d_in[0];
  const float* centers = (const float*)d_in[1];
  int*         out     = (int*)d_out;   // argmin -> int32 ids

  hipFuncSetAttribute((const void*)kmeans_assign_wmma,
                      hipFuncAttributeMaxDynamicSharedMemorySize, LDS_BYTES);
  dim3 grid(N_POINTS / M_TILE);   // 512 blocks
  dim3 block(256);                // 8 wave32
  kmeans_assign_wmma<<<grid, block, LDS_BYTES, stream>>>(x, centers, out);
}
